// Transformer_51024211477192
// MI455X (gfx1250) — matrix operand order repaired and checked by reading, not compiled, not run on hardware
//
#include <hip/hip_runtime.h>

// ---------------------------------------------------------------------------
// CDNA5 (gfx1250) wave32 WMMA types
// ---------------------------------------------------------------------------
typedef __bf16 bf16;
typedef __attribute__((ext_vector_type(16))) __bf16 v16bf;
typedef __attribute__((ext_vector_type(8)))  __bf16 v8bf;
typedef __attribute__((ext_vector_type(8)))  float  v8f;

union ABFrag { v16bf v; v8bf h[2]; };

__device__ __forceinline__ bf16 f2bf(float f) {
  union { float f; unsigned u; } a; a.f = f;
  unsigned r = a.u + 0x7fffu + ((a.u >> 16) & 1u);   // RNE to bf16
  union { unsigned short s; bf16 b; } c; c.s = (unsigned short)(r >> 16);
  return c.b;
}

// ---------------------------------------------------------------------------
// Async global->LDS copy (GLOBAL_LOAD_ASYNC_TO_LDS_B128, ASYNCcnt tracked).
// Generic LDS pointer low 32 bits == LDS byte address (ISA 10.2 aperture math).
// Flip USE_ASYNC_LDS to 0 if the asm mnemonic needs fixing.
// ---------------------------------------------------------------------------
#define USE_ASYNC_LDS 1

__device__ __forceinline__ void async_ld16(bf16* l, const bf16* g) {
#if USE_ASYNC_LDS
  unsigned lo = (unsigned)(size_t)l;
  asm volatile("global_load_async_to_lds_b128 %0, %1, off"
               :: "v"(lo), "v"(g) : "memory");
#else
  *(v8bf*)l = *(const v8bf*)g;
#endif
}
__device__ __forceinline__ void wait_async_le4() {
#if USE_ASYNC_LDS
  asm volatile("s_wait_asynccnt 0x4" ::: "memory");
#endif
}
__device__ __forceinline__ void wait_async_0() {
#if USE_ASYNC_LDS
  asm volatile("s_wait_asynccnt 0x0" ::: "memory");
#endif
}

// ---------------------------------------------------------------------------
// f32 -> bf16 bulk convert (n % 4 == 0)
// ---------------------------------------------------------------------------
__global__ __launch_bounds__(256)
void f32_to_bf16(const float* __restrict__ src, bf16* __restrict__ dst, int n) {
  const int base = (blockIdx.x * 256 + threadIdx.x) * 4;
  if (base < n) {
    float4 f = *(const float4*)(src + base);
    dst[base + 0] = f2bf(f.x); dst[base + 1] = f2bf(f.y);
    dst[base + 2] = f2bf(f.z); dst[base + 3] = f2bf(f.w);
  }
}

// ---------------------------------------------------------------------------
// C[M,N] = act(A[M,K] @ W[N,K]^T + bias); A,W bf16; f32 accumulate.
// Double-buffered LDS, async staging. Block 256 thr = 8 waves, tile 128x128,
// per-wave 32x64 (2x4 16x16 accumulators). M%128==0, N%128==0, K%32==0.
// OUTF: write f32 C, OUTB: write bf16 C.
// ---------------------------------------------------------------------------
template <int ACT, int OUTF, int OUTB>
__global__ __launch_bounds__(256)
void wmma_gemm(const bf16* __restrict__ A, const bf16* __restrict__ W,
               const float* __restrict__ bias,
               float* __restrict__ Cf, bf16* __restrict__ Cb,
               int M, int N, int K) {
  constexpr int LDT = 40;               // padded K-stride in LDS (bf16)
  __shared__ bf16 As[2][128 * LDT];
  __shared__ bf16 Bs[2][128 * LDT];

  const int tid   = threadIdx.x;
  const int lane  = tid & 31;
  const int wid   = tid >> 5;
  const int waveM = wid >> 1;           // 0..3 -> 32-row slice
  const int waveN = wid & 1;            // 0..1 -> 64-col slice
  const int half  = lane >> 4;
  const int l15   = lane & 15;

  const int blockM = blockIdx.y * 128;
  const int blockN = blockIdx.x * 128;

  const int srow = tid >> 1;            // staging row 0..127
  const int skb  = (tid & 1) * 16;      // 16-element half of the 32-wide K tile

  const bf16* Ag = A + (size_t)(blockM + srow) * K + skb;
  const bf16* Wg = W + (size_t)(blockN + srow) * K + skb;

  v8f acc[2][4] = {};

  auto stage = [&](int kt, int b) {
    const int k0 = kt << 5;
    bf16* la = &As[b][srow * LDT + skb];
    async_ld16(la,     Ag + k0);
    async_ld16(la + 8, Ag + k0 + 8);
    bf16* lb = &Bs[b][srow * LDT + skb];
    async_ld16(lb,     Wg + k0);
    async_ld16(lb + 8, Wg + k0 + 8);
  };

  const int KT = K >> 5;
  stage(0, 0);
  for (int kt = 0; kt < KT; ++kt) {
    const int cur = kt & 1;
    if (kt + 1 < KT) { stage(kt + 1, cur ^ 1); wait_async_le4(); }
    else             { wait_async_0(); }
    __syncthreads();                    // all waves' tile `cur` staged

    ABFrag af[2];
#pragma unroll
    for (int i = 0; i < 2; ++i) {
      const bf16* p = &As[cur][(waveM * 32 + i * 16 + l15) * LDT];
      af[i].h[0] = *(const v8bf*)(p + half * 16);
      af[i].h[1] = *(const v8bf*)(p + half * 16 + 8);
    }
    ABFrag bfm[4];
#pragma unroll
    for (int j = 0; j < 4; ++j) {
      const bf16* p = &Bs[cur][(waveN * 64 + j * 16 + l15) * LDT + half * 16];
      bfm[j].h[0] = *(const v8bf*)(p);
      bfm[j].h[1] = *(const v8bf*)(p + 8);
    }
#pragma unroll
    for (int i = 0; i < 2; ++i)
#pragma unroll
      for (int j = 0; j < 4; ++j)
        acc[i][j] = __builtin_amdgcn_wmma_f32_16x16x32_bf16(
            false, af[i].v, false, bfm[j].v, (short)0, acc[i][j], false, false);
    __syncthreads();                    // reads done before buffer reuse
  }

#pragma unroll
  for (int i = 0; i < 2; ++i) {
#pragma unroll
    for (int j = 0; j < 4; ++j) {
      const int col = blockN + waveN * 64 + j * 16 + l15;
      const float bv = bias[col];
#pragma unroll
      for (int v = 0; v < 8; ++v) {
        const int row = blockM + waveM * 32 + i * 16 + v + half * 8;
        float x = acc[i][j][v] + bv;
        if (ACT == 1) x = fmaxf(x, 0.0f);
        if (OUTF) Cf[(size_t)row * N + col] = x;
        if (OUTB) Cb[(size_t)row * N + col] = f2bf(x);
      }
    }
  }
}

// ---------------------------------------------------------------------------
// Flash-style attention.  qkv rows (bf16): [B*S][1536] = Q|K|V.  Additive
// band mask (+1 inside band), softmax over all S keys, ctx (bf16)[B*S][512].
// Grid: (S/32, B).  Block 256 thr = 8 waves: waveQ(2) x waveD(4).
// ---------------------------------------------------------------------------
__global__ __launch_bounds__(256)
void flash_attn(const bf16* __restrict__ qkv, bf16* __restrict__ ctx,
                const int* __restrict__ num_attn_p, int S) {
  constexpr int D = 512, QLD = 1536;
  constexpr int LQ = 520, LK = 520, LV = 40, LP = 40;

  __shared__ bf16  Qs[32 * LQ];
  __shared__ bf16  Ks[32 * LK];
  __shared__ bf16  Vts[512 * LV];       // transposed V chunk: [d][key]
  __shared__ float Sf[32 * 32];
  __shared__ bf16  Pb[32 * LP];
  __shared__ float mrowS[32], lrowS[32], arowS[32];

  const int b     = blockIdx.y;
  const int qbase = blockIdx.x * 32;
  const int na    = num_attn_p[0];
  const float scale = rsqrtf((float)D);

  const int tid  = threadIdx.x;
  const int lane = tid & 31, wid = tid >> 5;
  const int waveQ = wid >> 2;           // 0..1
  const int waveD = wid & 3;            // 0..3
  const int half = lane >> 4, l15 = lane & 15;

  const int srow = tid >> 3, scb = (tid & 7) * 64;

  // --- stage Q tile (32 x 512, async) ---
  {
    const bf16* qg = qkv + (size_t)(b * S + qbase + srow) * QLD + scb;
    bf16* qs = &Qs[srow * LQ + scb];
#pragma unroll
    for (int i = 0; i < 8; ++i) async_ld16(qs + i * 8, qg + i * 8);
  }
  if (tid < 32) { mrowS[tid] = -1e30f; lrowS[tid] = 0.0f; }

  v8f acc[8] = {};

  for (int kc = 0; kc < S / 32; ++kc) {
    // --- stage K chunk (async) and transposed V chunk (scalar scatter) ---
    {
      const size_t gbase = (size_t)(b * S + kc * 32 + srow) * QLD;
      const bf16* kg = qkv + gbase + 512 + scb;
      bf16* ks = &Ks[srow * LK + scb];
#pragma unroll
      for (int i = 0; i < 8; ++i) async_ld16(ks + i * 8, kg + i * 8);
      const bf16* vg = qkv + gbase + 1024 + scb;
#pragma unroll
      for (int i = 0; i < 8; ++i) {
        v8bf f = *(const v8bf*)(vg + i * 8);
#pragma unroll
        for (int j = 0; j < 8; ++j) Vts[(scb + i * 8 + j) * LV + srow] = f[j];
      }
    }
    wait_async_0();
    __syncthreads();

    // --- scores: waves 0..3 compute 32x32 (4 tiles of 16x16), K=512 ---
    if (wid < 4) {
      const int mq = wid >> 1, nk = wid & 1;
      v8f sacc = {};
#pragma unroll
      for (int kk = 0; kk < D / 32; ++kk) {
        ABFrag a, bb;
        const bf16* ap = &Qs[(mq * 16 + l15) * LQ + kk * 32];
        a.h[0] = *(const v8bf*)(ap + half * 16);
        a.h[1] = *(const v8bf*)(ap + half * 16 + 8);
        const bf16* bp = &Ks[(nk * 16 + l15) * LK + kk * 32 + half * 16];
        bb.h[0] = *(const v8bf*)(bp);
        bb.h[1] = *(const v8bf*)(bp + 8);
        sacc = __builtin_amdgcn_wmma_f32_16x16x32_bf16(
            false, a.v, false, bb.v, (short)0, sacc, false, false);
      }
      const int kj = kc * 32 + nk * 16 + l15;
#pragma unroll
      for (int v = 0; v < 8; ++v) {
        const int qi = qbase + mq * 16 + v + half * 8;
        const float mval = (kj <= qi && (qi - kj) < na) ? 1.0f : 0.0f;
        Sf[(mq * 16 + v + half * 8) * 32 + nk * 16 + l15] = sacc[v] * scale + mval;
      }
    }
    __syncthreads();

    // --- online softmax update (one thread per query row) ---
    if (tid < 32) {
      const int q = tid;
      float mo = mrowS[q], cm = mo;
#pragma unroll 8
      for (int j = 0; j < 32; ++j) cm = fmaxf(cm, Sf[q * 32 + j]);
      const float al = __expf(mo - cm);
      float s = 0.0f;
#pragma unroll 8
      for (int j = 0; j < 32; ++j) {
        const float p = __expf(Sf[q * 32 + j] - cm);
        s += p;
        Pb[q * LP + j] = f2bf(p);
      }
      mrowS[q] = cm;
      lrowS[q] = lrowS[q] * al + s;
      arowS[q] = al;
    }
    __syncthreads();

    // --- ctx = ctx*alpha + P @ Vchunk  (K=32: one WMMA per 16-col tile) ---
    ABFrag pa;
    {
      const bf16* pp = &Pb[(waveQ * 16 + l15) * LP];
      pa.h[0] = *(const v8bf*)(pp + half * 16);
      pa.h[1] = *(const v8bf*)(pp + half * 16 + 8);
    }
    float al[8];
#pragma unroll
    for (int v = 0; v < 8; ++v) al[v] = arowS[waveQ * 16 + v + half * 8];
#pragma unroll
    for (int wn = 0; wn < 8; ++wn) {
#pragma unroll
      for (int v = 0; v < 8; ++v) acc[wn][v] *= al[v];
      ABFrag vb;
      const bf16* vp = &Vts[(waveD * 128 + wn * 16 + l15) * LV + half * 16];
      vb.h[0] = *(const v8bf*)(vp);
      vb.h[1] = *(const v8bf*)(vp + 8);
      acc[wn] = __builtin_amdgcn_wmma_f32_16x16x32_bf16(
          false, pa.v, false, vb.v, (short)0, acc[wn], false, false);
    }
    __syncthreads();
  }

  // --- normalize by l, write bf16 ctx ---
  float linv[8];
#pragma unroll
  for (int v = 0; v < 8; ++v) linv[v] = 1.0f / lrowS[waveQ * 16 + v + half * 8];
#pragma unroll
  for (int wn = 0; wn < 8; ++wn) {
#pragma unroll
    for (int v = 0; v < 8; ++v) {
      const int q = qbase + waveQ * 16 + v + half * 8;
      const int d = waveD * 128 + wn * 16 + l15;
      ctx[(size_t)(b * S + q) * D + d] = f2bf(acc[wn][v] * linv[v]);
    }
  }
}

// ---------------------------------------------------------------------------
// outf/outb = LayerNorm(x + r) * g + beta   (one block per row, D = 512)
// ---------------------------------------------------------------------------
__global__ __launch_bounds__(256)
void residual_ln(const float* __restrict__ x, const float* __restrict__ r,
                 const float* __restrict__ g, const float* __restrict__ beta,
                 float* __restrict__ outf, bf16* __restrict__ outb, int D) {
  __shared__ float red[256];
  __shared__ float s_mu, s_rstd;
  const int row = blockIdx.x, tid = threadIdx.x;
  const size_t base = (size_t)row * D;
  const float e0 = x[base + tid] + r[base + tid];
  const float e1 = x[base + tid + 256] + r[base + tid + 256];
  red[tid] = e0 + e1;
  __syncthreads();
  for (int s = 128; s > 0; s >>= 1) { if (tid < s) red[tid] += red[tid + s]; __syncthreads(); }
  if (tid == 0) s_mu = red[0] / (float)D;
  __syncthreads();
  const float mu = s_mu;
  const float d0 = e0 - mu, d1 = e1 - mu;
  red[tid] = d0 * d0 + d1 * d1;
  __syncthreads();
  for (int s = 128; s > 0; s >>= 1) { if (tid < s) red[tid] += red[tid + s]; __syncthreads(); }
  if (tid == 0) s_rstd = rsqrtf(red[0] / (float)D + 1e-5f);
  __syncthreads();
  const float rs = s_rstd;
  const float o0 = d0 * rs * g[tid] + beta[tid];
  const float o1 = d1 * rs * g[tid + 256] + beta[tid + 256];
  outf[base + tid]       = o0;  outb[base + tid]       = f2bf(o0);
  outf[base + tid + 256] = o1;  outb[base + tid + 256] = f2bf(o1);
}

// ---------------------------------------------------------------------------
// Deterministic two-pass BatchNorm over 16384 rows x 256 cols
// ---------------------------------------------------------------------------
__global__ void bn_partial(const float* __restrict__ y, float* __restrict__ partS,
                           float* __restrict__ partQ, int rows_per_block) {
  const int col = threadIdx.x;
  const int r0 = blockIdx.x * rows_per_block;
  float s = 0.0f, q = 0.0f;
  for (int r = 0; r < rows_per_block; ++r) {
    const float v = y[(size_t)(r0 + r) * 256 + col];
    s += v; q += v * v;
  }
  partS[blockIdx.x * 256 + col] = s;
  partQ[blockIdx.x * 256 + col] = q;
}

__global__ void bn_reduce(const float* __restrict__ partS, const float* __restrict__ partQ,
                          float* __restrict__ stats, int nblocks) {
  const int col = threadIdx.x;
  float s = 0.0f, q = 0.0f;
  for (int i = 0; i < nblocks; ++i) { s += partS[i * 256 + col]; q += partQ[i * 256 + col]; }
  stats[col] = s; stats[256 + col] = q;
}

__global__ void bn_apply(const float* __restrict__ y, const float* __restrict__ stats,
                         const float* __restrict__ g, const float* __restrict__ b,
                         float* __restrict__ out, int nrows) {
  const int col = threadIdx.x, row = blockIdx.x;
  const float inv = 1.0f / (float)nrows;
  const float mu = stats[col] * inv;
  const float var = stats[256 + col] * inv - mu * mu;
  float v = (y[(size_t)row * 256 + col] - mu) * rsqrtf(var + 1e-5f) * g[col] + b[col];
  out[(size_t)row * 256 + col] = (v >= 0.0f) ? v : 0.02f * v;
}

// ---------------------------------------------------------------------------
extern "C" void kernel_launch(void* const* d_in, const int* in_sizes, int n_in,
                              void* d_out, int out_size, void* d_ws, size_t ws_size,
                              hipStream_t stream) {
  (void)in_sizes; (void)n_in; (void)out_size; (void)ws_size;
  const float* feature = (const float*)d_in[0];
  const float* Wqkv    = (const float*)d_in[1];
  const float* bqkv    = (const float*)d_in[2];
  const float* Wo      = (const float*)d_in[3];
  const float* bo      = (const float*)d_in[4];
  const float* W1      = (const float*)d_in[5];
  const float* b1      = (const float*)d_in[6];
  const float* W2      = (const float*)d_in[7];
  const float* b2      = (const float*)d_in[8];
  const float* ln1_g   = (const float*)d_in[9];
  const float* ln1_b   = (const float*)d_in[10];
  const float* ln2_g   = (const float*)d_in[11];
  const float* ln2_b   = (const float*)d_in[12];
  const float* Wl      = (const float*)d_in[13];
  const float* bl      = (const float*)d_in[14];
  const float* bn_g    = (const float*)d_in[15];
  const float* bn_b    = (const float*)d_in[16];
  const int*   num_attn= (const int*)d_in[17];
  float* out = (float*)d_out;

  const int B = 16, S = 1024, D = 512, DFF = 2048, DOUT = 256;
  const int M = B * S;                          // 16384

  char* p = (char*)d_ws;
  bf16* feat_bf = (bf16*)p;  p += (size_t)M * D * 2;
  bf16* wqkv_bf = (bf16*)p;  p += (size_t)1536 * 512 * 2;
  bf16* wo_bf   = (bf16*)p;  p += (size_t)512 * 512 * 2;
  bf16* w1_bf   = (bf16*)p;  p += (size_t)2048 * 512 * 2;
  bf16* w2_bf   = (bf16*)p;  p += (size_t)512 * 2048 * 2;
  bf16* wl_bf   = (bf16*)p;  p += (size_t)256 * 512 * 2;
  bf16* qkv_bf  = (bf16*)p;  p += (size_t)M * 1536 * 2;
  bf16* ctx_bf  = (bf16*)p;  p += (size_t)M * D * 2;
  float* attnf  = (float*)p; p += (size_t)M * D * 4;   // attnout, later x2f scratch
  float* x1f    = (float*)p; p += (size_t)M * D * 4;
  bf16* x1b     = (bf16*)p;  p += (size_t)M * D * 2;
  bf16* ff1_bf  = (bf16*)p;  p += (size_t)M * DFF * 2;
  float* ff2f   = (float*)p; p += (size_t)M * D * 4;
  bf16* x2b     = (bf16*)p;  p += (size_t)M * D * 2;
  float* yf     = (float*)p; p += (size_t)M * DOUT * 4;
  float* partS  = (float*)p; p += 256 * 256 * 4;
  float* partQ  = (float*)p; p += 256 * 256 * 4;
  float* stats  = (float*)p;

  dim3 blk(256);

  // 0) one-time (per launch) bf16 conversions of activations-in + weights
  f32_to_bf16<<<(M * D) / 1024, blk, 0, stream>>>(feature, feat_bf, M * D);
  f32_to_bf16<<<(1536 * 512) / 1024, blk, 0, stream>>>(Wqkv, wqkv_bf, 1536 * 512);
  f32_to_bf16<<<(512 * 512) / 1024, blk, 0, stream>>>(Wo, wo_bf, 512 * 512);
  f32_to_bf16<<<(2048 * 512) / 1024, blk, 0, stream>>>(W1, w1_bf, 2048 * 512);
  f32_to_bf16<<<(512 * 2048) / 1024, blk, 0, stream>>>(W2, w2_bf, 512 * 2048);
  f32_to_bf16<<<(256 * 512) / 1024, blk, 0, stream>>>(Wl, wl_bf, 256 * 512);

  // 1) qkv = feature @ Wqkv^T + bqkv   (bf16 out, feeds attention)
  wmma_gemm<0, 0, 1><<<dim3(1536 / 128, M / 128), blk, 0, stream>>>(
      feat_bf, wqkv_bf, bqkv, nullptr, qkv_bf, M, 1536, D);
  // 2) attention -> ctx (bf16)
  flash_attn<<<dim3(S / 32, B), blk, 0, stream>>>(qkv_bf, ctx_bf, num_attn, S);
  // 3) attnout = ctx @ Wo^T + bo (f32, feeds LN1)
  wmma_gemm<0, 1, 0><<<dim3(D / 128, M / 128), blk, 0, stream>>>(
      ctx_bf, wo_bf, bo, attnf, nullptr, M, D, D);
  // 4) x1 = LN1(feature + attnout)  (f32 + bf16)
  residual_ln<<<M, blk, 0, stream>>>(feature, attnf, ln1_g, ln1_b, x1f, x1b, D);
  // 5) ff1 = relu(x1 @ W1^T + b1) (bf16)
  wmma_gemm<1, 0, 1><<<dim3(DFF / 128, M / 128), blk, 0, stream>>>(
      x1b, w1_bf, b1, nullptr, ff1_bf, M, DFF, D);
  // 6) ff2 = ff1 @ W2^T + b2 (f32)
  wmma_gemm<0, 1, 0><<<dim3(D / 128, M / 128), blk, 0, stream>>>(
      ff1_bf, w2_bf, b2, ff2f, nullptr, M, D, DFF);
  // 7) x2 = LN2(x1 + ff2)  (bf16 used downstream; f32 copy to scratch)
  residual_ln<<<M, blk, 0, stream>>>(x1f, ff2f, ln2_g, ln2_b, attnf, x2b, D);
  // 8) y = x2 @ Wl^T + bl (f32)
  wmma_gemm<0, 1, 0><<<dim3(DOUT / 128, M / 128), blk, 0, stream>>>(
      x2b, wl_bf, bl, yf, nullptr, M, DOUT, D);
  // 9) BatchNorm (deterministic two-pass) + LeakyReLU(0.02)
  bn_partial<<<256, blk, 0, stream>>>(yf, partS, partQ, M / 256);
  bn_reduce<<<1, blk, 0, stream>>>(partS, partQ, stats, 256);
  bn_apply<<<M, blk, 0, stream>>>(yf, stats, bn_g, bn_b, out, M);
}
